// DAWN_14886356647950
// MI455X (gfx1250) — compile-verified, matrix operand order checked by reading, not run on hardware
//
#include <hip/hip_runtime.h>
#include <math.h>

// ---------------------------------------------------------------------------
// Problem constants (match reference: B=2, S=1024, D=1024, N=8192)
// ---------------------------------------------------------------------------
#define BS_   2048   // B*S rows
#define DD_   1024   // feature dim D
#define NT_   8192   // embedding pool N

typedef __attribute__((ext_vector_type(16))) __bf16 v16bf;
typedef __attribute__((ext_vector_type(8)))  __bf16 v8bf;
typedef __attribute__((ext_vector_type(8)))  float  v8f;

// Explicit global (addrspace 1) pointers so fragment loads lower to
// global_load_b128 (not flat_load_b128, which also ties up DScnt).
typedef const __attribute__((address_space(1))) __bf16 gbf16;
typedef const __attribute__((address_space(1))) v8bf   gv8bf;

__device__ __forceinline__ gbf16* as_global(const __bf16* p) {
  return (gbf16*)(unsigned long long)p;
}

union FragAB { v16bf v; v8bf h[2]; };

// NT fragment load for v_wmma_f32_16x16x32_bf16.
// 16-bit A/B operand layout (ISA 7.12.2): lanes 0-15 hold K {0..7,16..23},
// lanes 16-31 hold K {8..15,24..31}; two contiguous 16B chunks per lane.
// p already includes the per-lane row and +hi*8 offset.
__device__ __forceinline__ v16bf frag_ld(gbf16* p) {
  FragAB f;
  f.h[0] = ((gv8bf*)p)[0];   // K = k + hi*8 .. +7
  f.h[1] = ((gv8bf*)p)[2];   // K = k + 16 + hi*8 .. +7
  return f.v;
}

__device__ __forceinline__ void wmma8(v8f (&acc)[2][4],
                                      const v16bf (&af)[2],
                                      const v16bf (&bf)[4]) {
  #pragma unroll
  for (int t = 0; t < 2; ++t)
    #pragma unroll
    for (int u = 0; u < 4; ++u)
      acc[t][u] = __builtin_amdgcn_wmma_f32_16x16x32_bf16(
          false, af[t], false, bf[u], (short)0, acc[t][u], false, false);
}

// ---------------------------------------------------------------------------
// Generic 128x128-tile NT GEMM: C-ish = A[M,K] @ B[N,K]^T, fp32 accum.
// 256 threads = 8 waves in a 4(M) x 2(N) grid; 2x4 wmma tiles per wave.
// K-loop ping-pongs two fragment sets (K-step 64, K % 64 == 0 required):
// loads for one set issue while the other set's 8 WMMAs run — no register
// rotation copies, L2 latency overlaps the XDL pipe.
// MODE 0: store bf16(acc)                          (sc0; masked to chunk 0)
// MODE 1: gate epilogue -> store eg bf16           (needs tau rowvec)
// MODE 2: in-place eg *= bf16(acc)                 (egxr)
// MODE 3: store fp32 acc * scale[m] to d_out       (needs scale rowvec)
// ---------------------------------------------------------------------------
template<int MODE>
__launch_bounds__(256)
__global__ void gemm_nt_kernel(const __bf16* __restrict__ A,
                               const __bf16* __restrict__ B,
                               void* __restrict__ Cv,
                               int K, int lda, int ldb, int ldc,
                               const float* __restrict__ rowvec,
                               const int* __restrict__ ncp, int Ntot) {
  const int n0 = blockIdx.x * 128;
  const int m0 = blockIdx.y * 128;
  if constexpr (MODE == 0) {
    int ncv = *ncp; if (ncv < 1) ncv = 1;
    const int cs = Ntot / ncv;          // chunk-0 width
    if (n0 >= cs) return;               // uniform per-block exit
  }
  const int tid  = threadIdx.x;
  const int wave = tid >> 5;
  const int lane = tid & 31;
  const int l15  = lane & 15;
  const int hi   = lane >> 4;
  const int wm   = wave & 3;            // 4 waves along M
  const int wn   = wave >> 2;           // 2 waves along N
  const int mBase = m0 + wm * 32;       // 2 m-tiles of 16
  const int nBase = n0 + wn * 64;       // 4 n-tiles of 16

  // Per-lane fragment pointers (global addrspace); bumped 64 elems per K-step.
  gbf16* ap[2];
  gbf16* bp[4];
  #pragma unroll
  for (int t = 0; t < 2; ++t)
    ap[t] = as_global(A) + (size_t)(mBase + t * 16 + l15) * lda + hi * 8;
  #pragma unroll
  for (int u = 0; u < 4; ++u)
    bp[u] = as_global(B) + (size_t)(nBase + u * 16 + l15) * ldb + hi * 8;

  v8f acc[2][4];
  const v8f vzero = {0.f,0.f,0.f,0.f,0.f,0.f,0.f,0.f};
  #pragma unroll
  for (int t = 0; t < 2; ++t)
    #pragma unroll
    for (int u = 0; u < 4; ++u) acc[t][u] = vzero;

  v16bf a0[2], b0[4], a1[2], b1[4];
  // set0 @ k = 0
  #pragma unroll
  for (int t = 0; t < 2; ++t) a0[t] = frag_ld(ap[t]);
  #pragma unroll
  for (int u = 0; u < 4; ++u) b0[u] = frag_ld(bp[u]);

  int k = 0;
  for (; k + 64 < K; k += 64) {
    // set1 @ k+32
    #pragma unroll
    for (int t = 0; t < 2; ++t) a1[t] = frag_ld(ap[t] + 32);
    #pragma unroll
    for (int u = 0; u < 4; ++u) b1[u] = frag_ld(bp[u] + 32);
    wmma8(acc, a0, b0);
    #pragma unroll
    for (int t = 0; t < 2; ++t) ap[t] += 64;
    #pragma unroll
    for (int u = 0; u < 4; ++u) bp[u] += 64;
    // set0 @ k+64
    #pragma unroll
    for (int t = 0; t < 2; ++t) a0[t] = frag_ld(ap[t]);
    #pragma unroll
    for (int u = 0; u < 4; ++u) b0[u] = frag_ld(bp[u]);
    wmma8(acc, a1, b1);
  }
  // tail (k + 64 == K): set1 @ k+32, then both octets.
  #pragma unroll
  for (int t = 0; t < 2; ++t) a1[t] = frag_ld(ap[t] + 32);
  #pragma unroll
  for (int u = 0; u < 4; ++u) b1[u] = frag_ld(bp[u] + 32);
  wmma8(acc, a0, b0);
  wmma8(acc, a1, b1);

  // D layout (ISA 7.12.2): element (m = 8*hi + v, n = l15) in vgpr v.
  #pragma unroll
  for (int t = 0; t < 2; ++t) {
    #pragma unroll
    for (int u = 0; u < 4; ++u) {
      #pragma unroll
      for (int v = 0; v < 8; ++v) {
        const int m = mBase + t * 16 + hi * 8 + v;
        const int n = nBase + u * 16 + l15;
        const size_t idx = (size_t)m * ldc + n;
        const float val = acc[t][u][v];
        if constexpr (MODE == 0) {
          ((__bf16*)Cv)[idx] = (__bf16)val;
        } else if constexpr (MODE == 1) {
          // raw = bf16(sc) - bf16(tau) in bf16
          const float scb  = (float)(__bf16)val;
          const float taub = (float)(__bf16)rowvec[m];
          const __bf16 raw_b = (__bf16)(scb - taub);
          const float rf = (float)raw_b;
          __bf16 gc;
          if (rf > 0.f) gc = raw_b;
          else          gc = (__bf16)(1e-6f * expf(fmaxf(fminf(rf, 0.f), -10.f)));
          float gcf = (float)gc;
          gcf = fminf(fmaxf(gcf, 0.f), 10.f);
          const float ef = expf(gcf) - 1.f;
          ((__bf16*)Cv)[idx] = (__bf16)ef;          // eg
        } else if constexpr (MODE == 2) {
          __bf16* C = (__bf16*)Cv;
          const float e   = (float)C[idx];          // eg
          const float xrb = (float)(__bf16)val;     // bf16(xr)
          C[idx] = (__bf16)(e * xrb);               // egxr (in place)
        } else {
          ((float*)Cv)[idx] = val * rowvec[m];      // out = raw_out*inv_es*gs
        }
      }
    }
  }
}

// ---------------------------------------------------------------------------
// Prep kernels
// ---------------------------------------------------------------------------
__global__ void cvt_kernel(const float* __restrict__ src, __bf16* __restrict__ dst, int n) {
  int i = blockIdx.x * blockDim.x + threadIdx.x;
  const int stride = gridDim.x * blockDim.x;
  for (; i < n; i += stride) dst[i] = (__bf16)src[i];
}

// emb row-normalize (fp32 norm) -> bf16
__global__ void norm_rows_kernel(const float* __restrict__ emb, __bf16* __restrict__ dst, int D) {
  __shared__ float red[256];
  __shared__ float invn;
  const int r = blockIdx.x, tid = threadIdx.x;
  const float* row = emb + (size_t)r * D;
  float s = 0.f;
  for (int j = tid; j < D; j += 256) { const float v = row[j]; s += v * v; }
  red[tid] = s; __syncthreads();
  for (int off = 128; off > 0; off >>= 1) {
    if (tid < off) red[tid] += red[tid + off];
    __syncthreads();
  }
  if (tid == 0) invn = 1.f / (sqrtf(red[0]) + 1e-8f);
  __syncthreads();
  const float iv = invn;
  for (int j = tid; j < D; j += 256) dst[(size_t)r * D + j] = (__bf16)(row[j] * iv);
}

// w_write [N,D] fp32 -> writeT [D,N] bf16 (tiled transpose)
__global__ void tcvt_kernel(const float* __restrict__ src, __bf16* __restrict__ dst,
                            int rows /*N*/, int cols /*D*/) {
  __shared__ __bf16 tile[32][33];
  const int dBase = blockIdx.x * 32;
  const int nBase = blockIdx.y * 32;
  const int tx = threadIdx.x, ty = threadIdx.y;
  for (int i = ty; i < 32; i += 8)
    tile[i][tx] = (__bf16)src[(size_t)(nBase + i) * cols + dBase + tx];
  __syncthreads();
  for (int i = ty; i < 32; i += 8)
    dst[(size_t)(dBase + i) * rows + nBase + tx] = tile[tx][i];
}

// ---------------------------------------------------------------------------
// Row reductions
// ---------------------------------------------------------------------------
__global__ void row_tau_kernel(const __bf16* __restrict__ sc0,
                               const float* __restrict__ tau_off,
                               float* __restrict__ tau,
                               const int* __restrict__ ncp, int Ntot, int ld) {
  __shared__ float ss[256], sq[256];
  const int m = blockIdx.x, tid = threadIdx.x;
  int nc = *ncp; if (nc < 1) nc = 1;
  const int cs = Ntot / nc;
  float s = 0.f, q = 0.f;
  for (int j = tid; j < cs; j += 256) {
    const float v = (float)sc0[(size_t)m * ld + j];
    s += v;
    q += (float)(__bf16)(v * v);   // sc0**2 in bf16, summed fp32
  }
  ss[tid] = s; sq[tid] = q; __syncthreads();
  for (int off = 128; off > 0; off >>= 1) {
    if (tid < off) { ss[tid] += ss[tid + off]; sq[tid] += sq[tid + off]; }
    __syncthreads();
  }
  if (tid == 0) {
    const float s_sum  = ss[0] * (float)nc;
    const float sq_sum = sq[0] * (float)nc;
    const float mean = s_sum / (float)Ntot;
    const float var  = sq_sum / (float)Ntot - mean * mean;
    const float stdv = sqrtf(fmaxf(var, 0.f)) + 1e-8f;
    tau[m] = mean + tau_off[m] * stdv;
  }
}

// scale[m] = float(bf16(1/(tes+1e-8))) * float(bf16(tanh(tem)))
__global__ void row_scale_kernel(const __bf16* __restrict__ eg,
                                 float* __restrict__ scale, int Ntot, int ld) {
  __shared__ float ss[256], sm[256];
  const int m = blockIdx.x, tid = threadIdx.x;
  float s = 0.f, mx = 0.f;     // eg >= 0 always, so 0 is a safe max-identity
  for (int j = tid; j < Ntot; j += 256) {
    const float v = (float)eg[(size_t)m * ld + j];
    s += v;
    mx = fmaxf(mx, v);
  }
  ss[tid] = s; sm[tid] = mx; __syncthreads();
  for (int off = 128; off > 0; off >>= 1) {
    if (tid < off) { ss[tid] += ss[tid + off]; sm[tid] = fmaxf(sm[tid], sm[tid + off]); }
    __syncthreads();
  }
  if (tid == 0) {
    const float inv = (float)(__bf16)(1.f / (ss[0] + 1e-8f));
    const float gs  = (float)(__bf16)tanhf(sm[0]);
    scale[m] = inv * gs;
  }
}

// ---------------------------------------------------------------------------
// Host launcher
// ---------------------------------------------------------------------------
extern "C" void kernel_launch(void* const* d_in, const int* in_sizes, int n_in,
                              void* d_out, int out_size, void* d_ws, size_t ws_size,
                              hipStream_t stream) {
  (void)in_sizes; (void)n_in; (void)out_size; (void)ws_size;
  const float* x       = (const float*)d_in[0];
  const float* h       = (const float*)d_in[1];
  const float* emb     = (const float*)d_in[2];
  const float* tau_off = (const float*)d_in[3];
  const float* w_read  = (const float*)d_in[4];
  const float* w_write = (const float*)d_in[5];
  const int*   ncp     = (const int*)d_in[6];

  char* ws = (char*)d_ws;
  size_t o = 0;
  __bf16* h_bf    = (__bf16*)(ws + o); o += (size_t)BS_ * DD_ * 2;   //  4 MB
  __bf16* x_bf    = (__bf16*)(ws + o); o += (size_t)BS_ * DD_ * 2;   //  4 MB
  __bf16* embN    = (__bf16*)(ws + o); o += (size_t)NT_ * DD_ * 2;   // 16 MB
  __bf16* read_bf = (__bf16*)(ws + o); o += (size_t)NT_ * DD_ * 2;   // 16 MB
  __bf16* writeT  = (__bf16*)(ws + o); o += (size_t)NT_ * DD_ * 2;   // 16 MB
  __bf16* egbuf   = (__bf16*)(ws + o); o += (size_t)BS_ * NT_ * 2;   // 32 MB
  float*  tau     = (float*)(ws + o);  o += (size_t)BS_ * 4;
  float*  scale   = (float*)(ws + o);  o += (size_t)BS_ * 4;

  // --- prep ---
  cvt_kernel<<<2048, 256, 0, stream>>>(h, h_bf, BS_ * DD_);
  cvt_kernel<<<2048, 256, 0, stream>>>(x, x_bf, BS_ * DD_);
  cvt_kernel<<<4096, 256, 0, stream>>>(w_read, read_bf, NT_ * DD_);
  norm_rows_kernel<<<NT_, 256, 0, stream>>>(emb, embN, DD_);
  tcvt_kernel<<<dim3(DD_ / 32, NT_ / 32), dim3(32, 8), 0, stream>>>(w_write, writeT, NT_, DD_);

  const dim3 gFull(NT_ / 128, BS_ / 128);   // N-major grid for [BS, N] outputs
  const dim3 gOut(DD_ / 128, BS_ / 128);    // for the [BS, D] output

  // --- sc0 (chunk 0 only, masked in-kernel via n_chunks) ---
  gemm_nt_kernel<0><<<gFull, 256, 0, stream>>>(h_bf, embN, egbuf,
      DD_, DD_, DD_, NT_, nullptr, ncp, NT_);
  row_tau_kernel<<<BS_, 256, 0, stream>>>(egbuf, tau_off, tau, ncp, NT_, NT_);

  // --- eg = gate(h @ embN^T - tau) over full N ---
  gemm_nt_kernel<1><<<gFull, 256, 0, stream>>>(h_bf, embN, egbuf,
      DD_, DD_, DD_, NT_, tau, nullptr, NT_);
  row_scale_kernel<<<BS_, 256, 0, stream>>>(egbuf, scale, NT_, NT_);

  // --- egxr = eg * (x @ w_read^T), in place ---
  gemm_nt_kernel<2><<<gFull, 256, 0, stream>>>(x_bf, read_bf, egbuf,
      DD_, DD_, DD_, NT_, nullptr, nullptr, NT_);

  // --- out = (egxr @ w_write) * scale[m], fp32 ---
  gemm_nt_kernel<3><<<gOut, 256, 0, stream>>>(egbuf, writeT, d_out,
      NT_, NT_, NT_, DD_, scale, nullptr, DD_);
}
